// MultiHeadSelfAttention_12781822673235
// MI455X (gfx1250) — matrix operand order repaired and checked by reading, not compiled, not run on hardware
//
#include <hip/hip_runtime.h>
#include <hip/hip_bf16.h>
#include <math.h>

// ---------------------------------------------------------------------------
// Multi-head self-attention (B=4, L=2048, D=1024, H=16, DH=64) for gfx1250.
// All matmuls run on v_wmma_f32_16x16x32_f16 (f16 inputs, f32 accumulate).
// Pipeline:
//   1) gemm_f16wmma<1>: qkv = x @ W^T, fused RoPE epilogue -> Qh/Kh/Vh f16
//   2) flash_attn: causal online-softmax attention, one wave per 16-q tile,
//      K/V tiles double-buffered in LDS via GLOBAL_LOAD_ASYNC_TO_LDS_B128
//   3) gemm_f16wmma<0>: out = O @ Wo^T (fp32 output)
// ---------------------------------------------------------------------------

typedef _Float16 half16 __attribute__((ext_vector_type(16)));
typedef float    float8 __attribute__((ext_vector_type(8)));

#define WMMA_F32_F16(a, b, c) \
  __builtin_amdgcn_wmma_f32_16x16x32_f16(false, (a), false, (b), (short)0, (c), false, false)

constexpr int Bc = 4, Lc = 2048, Dc = 1024, Hc = 16, DHc = 64;
constexpr float LOG_THETA = 11.512925464970229f; // ln(100000)

__device__ inline void cp16B(_Float16* d, const _Float16* s) {
  *reinterpret_cast<float4*>(d) = *reinterpret_cast<const float4*>(s);
}

// Async copy of one 128-byte row (global f16 -> LDS), 8 x b128, ASYNCcnt.
// lds_off: LDS byte offset (low 32 bits of the generic shared pointer);
// the instruction offset is applied to both the LDS and global address.
__device__ inline void async_row128(unsigned lds_off, const _Float16* g) {
  unsigned long long ga = (unsigned long long)(uintptr_t)g;
  asm volatile(
      "global_load_async_to_lds_b128 %0, %1, off\n\t"
      "global_load_async_to_lds_b128 %0, %1, off offset:16\n\t"
      "global_load_async_to_lds_b128 %0, %1, off offset:32\n\t"
      "global_load_async_to_lds_b128 %0, %1, off offset:48\n\t"
      "global_load_async_to_lds_b128 %0, %1, off offset:64\n\t"
      "global_load_async_to_lds_b128 %0, %1, off offset:80\n\t"
      "global_load_async_to_lds_b128 %0, %1, off offset:96\n\t"
      "global_load_async_to_lds_b128 %0, %1, off offset:112"
      :
      : "v"(lds_off), "v"(ga)
      : "memory");
}

__device__ inline void wait_async0() {
  asm volatile("s_wait_asynccnt 0x0" ::: "memory");
}

// ---------------------------------------------------------------------------
// C[M,N] = A[M,K] * Bm[N,K]^T   (both row-major, f32 in, f16 WMMA, f32 acc)
// MODE 0: plain f32 store to C.
// MODE 1: QKV epilogue: n -> (s,h,dh); apply RoPE to q/k; scatter f16 to
//         Qh/Kh/Vh laid out [B,H,L,DH].
// Block: 128 threads (4 waves). Block tile 64x64, K-step 32.
// ---------------------------------------------------------------------------
template <int MODE>
__global__ __launch_bounds__(128) void gemm_f16wmma(
    const float* __restrict__ A, const float* __restrict__ Bm,
    float* __restrict__ C, int M, int N, int K,
    _Float16* __restrict__ Qh, _Float16* __restrict__ Kh,
    _Float16* __restrict__ Vh) {
  __shared__ _Float16 As[64 * 32];  // [row][k]
  __shared__ _Float16 Bs[64 * 32];  // [n][k]

  const int tid = threadIdx.x;
  const int lane = tid & 31;
  const int wave = tid >> 5;
  const int hi = lane >> 4;
  const int ln = lane & 15;
  const int m0 = blockIdx.y * 64;
  const int n0 = blockIdx.x * 64;

  float8 acc[4];
#pragma unroll
  for (int t = 0; t < 4; ++t) acc[t] = (float8){};

  for (int kc = 0; kc < K; kc += 32) {
    // Cooperative stage: 2048 f32->f16 elements each for A and B tiles.
#pragma unroll
    for (int e = 0; e < 16; ++e) {
      int flat = e * 128 + tid;
      int r = flat >> 5, c = flat & 31;
      As[r * 32 + c] = (_Float16)A[(size_t)(m0 + r) * K + kc + c];
      Bs[r * 32 + c] = (_Float16)Bm[(size_t)(n0 + r) * K + kc + c];
    }
    __syncthreads();

    if (kc + 32 < K) {  // speculative prefetch of next K-slab (global_prefetch)
      __builtin_prefetch(&A[(size_t)(m0 + (tid >> 1)) * K + kc + 32], 0, 1);
      __builtin_prefetch(&Bm[(size_t)(n0 + (tid >> 1)) * K + kc + 32], 0, 1);
    }

    // A fragment: wave's 16-row strip, 16x32 f16.
    half16 af;
    {
      _Float16* ap = reinterpret_cast<_Float16*>(&af);
      const _Float16* arow = &As[(wave * 16 + ln) * 32];
      cp16B(ap, arow + hi * 16);          // K = hi*16 + 0..7
      cp16B(ap + 8, arow + hi * 16 + 8);  // K = hi*16 + 8..15
    }
#pragma unroll
    for (int t = 0; t < 4; ++t) {
      half16 bf;
      _Float16* bp = reinterpret_cast<_Float16*>(&bf);
      const _Float16* brow = &Bs[(t * 16 + ln) * 32];
      cp16B(bp, brow + hi * 16);          // K = hi*16 + 0..7
      cp16B(bp + 8, brow + hi * 16 + 8);  // K = hi*16 + 8..15
      acc[t] = WMMA_F32_F16(af, bf, acc[t]);
    }
    __syncthreads();
  }

  if (MODE == 0) {
#pragma unroll
    for (int t = 0; t < 4; ++t) {
      int n = n0 + t * 16 + ln;
#pragma unroll
      for (int e = 0; e < 8; ++e) {
        int m = m0 + wave * 16 + e + 8 * hi;
        C[(size_t)m * N + n] = acc[t][e];
      }
    }
  } else {
#pragma unroll
    for (int t = 0; t < 4; ++t) {
      int n = n0 + t * 16 + ln;
      int s = n >> 10;        // 0=q, 1=k, 2=v
      int rind = n & 1023;
      int h = rind >> 6;
      int dh = rind & 63;
      float invfreq = expf(-((float)(dh & ~1) / (float)DHc) * LOG_THETA);
#pragma unroll
      for (int e = 0; e < 8; ++e) {
        int m = m0 + wave * 16 + e + 8 * hi;
        int b = m >> 11;      // L = 2048
        int l = m & 2047;
        float v = acc[t][e];
        size_t dst = (((size_t)(b * Hc + h) * Lc) + l) * DHc + dh;
        if (s == 2) {
          Vh[dst] = (_Float16)v;
        } else {
          // RoPE pair lives in the adjacent lane (dh parity == lane parity).
          float partner = __shfl_xor(v, 1, 32);
          float ang = (float)l * invfreq;
          float sn, cs;
          sincosf(ang, &sn, &cs);
          float out = (dh & 1) ? (v * cs + partner * sn)
                               : (v * cs - partner * sn);
          if (s == 0) Qh[dst] = (_Float16)out;
          else        Kh[dst] = (_Float16)out;
        }
      }
    }
  }
}

// ---------------------------------------------------------------------------
// Causal flash attention. One wave32 per (b, h, 16-query tile).
// Q tile: 16x64 f16 = two 16x32 A-frags. 32-key blocks double-buffered in
// LDS via async-to-LDS copies (ASYNCcnt), overlapped with WMMA compute.
// S = Q K^T via WMMA; online softmax in C-fragment layout (row = vgpr+8*hi,
// col = lane&15); P goes through LDS to become an A-frag; O += P V via WMMA.
// ---------------------------------------------------------------------------
__global__ __launch_bounds__(32) void flash_attn(
    const _Float16* __restrict__ Qh, const _Float16* __restrict__ Kh,
    const _Float16* __restrict__ Vh, float* __restrict__ O) {
  __shared__ _Float16 Kt[2][32 * 64];
  __shared__ _Float16 Vt[2][32 * 64];
  __shared__ _Float16 Pt[16 * 32];

  const int lane = threadIdx.x;
  const int hi = lane >> 4;
  const int ln = lane & 15;
  const int q0 = blockIdx.x * 16;
  const int h = blockIdx.y;
  const int b = blockIdx.z;
  const size_t headoff = (size_t)(b * Hc + h) * Lc * DHc;

  // Q fragments.
  half16 aQ[2];
  {
    const _Float16* qrow = Qh + headoff + (size_t)(q0 + ln) * DHc;
#pragma unroll
    for (int c = 0; c < 2; ++c) {
      _Float16* ap = reinterpret_cast<_Float16*>(&aQ[c]);
      cp16B(ap, qrow + c * 32 + hi * 16);
      cp16B(ap + 8, qrow + c * 32 + hi * 16 + 8);
    }
  }

  float8 oacc[4];
#pragma unroll
  for (int c = 0; c < 4; ++c) oacc[c] = (float8){};
  float rowmax[8], rowsum[8];
#pragma unroll
  for (int e = 0; e < 8; ++e) { rowmax[e] = -__builtin_inff(); rowsum[e] = 0.0f; }

  const int nkb = (q0 + 16 + 31) >> 5;

  // Issue async staging of a 32-key K/V block into buffer `buf`.
  auto stage = [&](int kb, int buf) {
    const int krow = kb * 32 + lane;
    _Float16* kdst = &Kt[buf][lane * 64];
    _Float16* vdst = &Vt[buf][lane * 64];
    if (krow < Lc) {
      async_row128((unsigned)(uintptr_t)kdst, Kh + headoff + (size_t)krow * DHc);
      async_row128((unsigned)(uintptr_t)vdst, Vh + headoff + (size_t)krow * DHc);
    } else {
      float4 z{0.f, 0.f, 0.f, 0.f};
      float4* kd = reinterpret_cast<float4*>(kdst);
      float4* vd = reinterpret_cast<float4*>(vdst);
#pragma unroll
      for (int j = 0; j < 8; ++j) { kd[j] = z; vd[j] = z; }
    }
  };

  stage(0, 0);
  int buf = 0;
  for (int kb = 0; kb < nkb; ++kb) {
    const int key0 = kb * 32;
    wait_async0();     // drain async copies for this buffer
    __syncthreads();
    if (kb + 1 < nkb) stage(kb + 1, buf ^ 1);  // prefetch next block

    // S tiles for this key block (two 16x16 tiles).
    float8 sfr[2];
#pragma unroll
    for (int t = 0; t < 2; ++t) {
      float8 cacc = (float8){};
#pragma unroll
      for (int c = 0; c < 2; ++c) {
        half16 bf;
        _Float16* bp = reinterpret_cast<_Float16*>(&bf);
        const _Float16* kr = &Kt[buf][(t * 16 + ln) * 64 + c * 32 + hi * 16];
        cp16B(bp, kr);
        cp16B(bp + 8, kr + 8);
        cacc = WMMA_F32_F16(aQ[c], bf, cacc);
      }
      const int colg = key0 + t * 16 + ln;
#pragma unroll
      for (int e = 0; e < 8; ++e) {
        int rowg = q0 + e + 8 * hi;
        float sv = cacc[e] * 0.125f;  // 1/sqrt(64)
        sfr[t][e] = (colg <= rowg) ? sv : -__builtin_inff();
      }
    }

    // Online softmax update (16-lane reductions stay inside each half-wave).
    float alpha[8];
#pragma unroll
    for (int e = 0; e < 8; ++e) {
      float m = fmaxf(sfr[0][e], sfr[1][e]);
#pragma unroll
      for (int off = 8; off >= 1; off >>= 1)
        m = fmaxf(m, __shfl_xor(m, off, 32));
      float nm = fmaxf(rowmax[e], m);
      alpha[e] = expf(rowmax[e] - nm);
      rowmax[e] = nm;
      float p0 = expf(sfr[0][e] - nm);
      float p1 = expf(sfr[1][e] - nm);
      sfr[0][e] = p0;
      sfr[1][e] = p1;
      float ps = p0 + p1;
#pragma unroll
      for (int off = 8; off >= 1; off >>= 1)
        ps += __shfl_xor(ps, off, 32);
      rowsum[e] = rowsum[e] * alpha[e] + ps;
    }
#pragma unroll
    for (int c = 0; c < 4; ++c)
#pragma unroll
      for (int e = 0; e < 8; ++e) oacc[c][e] *= alpha[e];

    // P: C-layout -> LDS row-major [16][32] -> A-fragment.
#pragma unroll
    for (int t = 0; t < 2; ++t)
#pragma unroll
      for (int e = 0; e < 8; ++e)
        Pt[(e + 8 * hi) * 32 + t * 16 + ln] = (_Float16)sfr[t][e];
    __syncthreads();

    half16 aP;
    {
      _Float16* ap = reinterpret_cast<_Float16*>(&aP);
      const _Float16* prow = &Pt[ln * 32];
      cp16B(ap, prow + hi * 16);
      cp16B(ap + 8, prow + hi * 16 + 8);
    }
    // O += P * V over 4 dim-chunks of 16.
#pragma unroll
    for (int c = 0; c < 4; ++c) {
      half16 bf;
      _Float16* bp = reinterpret_cast<_Float16*>(&bf);
#pragma unroll
      for (int i = 0; i < 16; ++i)
        bp[i] = Vt[buf][(hi * 16 + i) * 64 + c * 16 + ln];
      oacc[c] = WMMA_F32_F16(aP, bf, oacc[c]);
    }
    __syncthreads();
    buf ^= 1;
  }

  // Normalize and store O as [B, L, D] fp32 (d = h*64 + chunk*16 + ln).
#pragma unroll
  for (int e = 0; e < 8; ++e) {
    float inv = 1.0f / rowsum[e];
    int l = q0 + e + 8 * hi;
    float* orow = O + ((size_t)b * Lc + l) * Dc + h * DHc;
#pragma unroll
    for (int c = 0; c < 4; ++c) orow[c * 16 + ln] = oacc[c][e] * inv;
  }
}

// ---------------------------------------------------------------------------
extern "C" void kernel_launch(void* const* d_in, const int* in_sizes, int n_in,
                              void* d_out, int out_size, void* d_ws,
                              size_t ws_size, hipStream_t stream) {
  const float* x  = (const float*)d_in[0];   // [B, L, D]
  const float* W  = (const float*)d_in[1];   // [3D, D]
  const float* Wo = (const float*)d_in[2];   // [D, D]
  float* out = (float*)d_out;                // [B, L, D]

  const size_t nElem = (size_t)Bc * Hc * Lc * DHc;  // 8,388,608
  _Float16* Qh = (_Float16*)d_ws;
  _Float16* Kh = Qh + nElem;
  _Float16* Vh = Kh + nElem;
  float* Ows = (float*)(Vh + nElem);  // [B, L, D] fp32 attention output

  const int M = Bc * Lc;  // 8192

  // 1) QKV projection + fused RoPE -> Qh/Kh/Vh
  gemm_f16wmma<1><<<dim3((3 * Dc) / 64, M / 64), dim3(128), 0, stream>>>(
      x, W, nullptr, M, 3 * Dc, Dc, Qh, Kh, Vh);

  // 2) Causal flash attention -> Ows
  flash_attn<<<dim3(Lc / 16, Hc, Bc), dim3(32), 0, stream>>>(Qh, Kh, Vh, Ows);

  // 3) Output projection -> d_out
  gemm_f16wmma<0><<<dim3(Dc / 64, M / 64), dim3(128), 0, stream>>>(
      Ows, Wo, out, M, Dc, Dc, nullptr, nullptr, nullptr);
}
